// MTR_Head1_18889266167857
// MI455X (gfx1250) — matrix operand order repaired and checked by reading, not compiled, not run on hardware
//
#include <hip/hip_runtime.h>
#include <cstdint>
#include <cstddef>

typedef __attribute__((ext_vector_type(16))) __bf16 v16bf;
typedef __attribute__((ext_vector_type(8)))  float  v8f;

#define B_       8
#define C_       256
#define HW_      4096
#define NP_      32768          // total pixels B*H*W
#define K_       2304           // C * 9 taps
#define OC_      512            // 256 loc + 256 cla output channels fused
#define CHUNK_PX 8192           // 2 batch images per chunk
#define NCHUNK   4

__device__ __forceinline__ unsigned short f2bf(float f) {
  unsigned u = __builtin_bit_cast(unsigned, f);
  u += 0x7FFFu + ((u >> 16) & 1u);          // round-to-nearest-even
  return (unsigned short)(u >> 16);
}
__device__ __forceinline__ float bf2f(unsigned short h) {
  return __builtin_bit_cast(float, ((unsigned)h) << 16);
}

// ---------------------------------------------------------------------------
// Kernel 1: pack deform-conv weights (loc||cla) to bf16 + fold BN params.
// w.reshape(O,C,9) contraction index k=c*9+n is exactly the inner flat order.
// ---------------------------------------------------------------------------
__global__ __launch_bounds__(256) void pack_prep(
    const float* __restrict__ w_dloc, const float* __restrict__ w_dcla,
    const float* __restrict__ g_loc,  const float* __restrict__ be_loc,
    const float* __restrict__ m_loc,  const float* __restrict__ v_loc,
    const float* __restrict__ g_cla,  const float* __restrict__ be_cla,
    const float* __restrict__ m_cla,  const float* __restrict__ v_cla,
    unsigned short* __restrict__ Wp, float* __restrict__ bns, float* __restrict__ bnb)
{
  int gid = blockIdx.x * 256 + threadIdx.x;
  if (gid < OC_ * K_) {
    int o = gid / K_;
    int k = gid - o * K_;
    float wv = (o < 256) ? w_dloc[(size_t)o * K_ + k]
                         : w_dcla[(size_t)(o - 256) * K_ + k];
    Wp[gid] = f2bf(wv);
  }
  if (gid < OC_) {
    float g, be, m, v;
    if (gid < 256) { g = g_loc[gid]; be = be_loc[gid]; m = m_loc[gid]; v = v_loc[gid]; }
    else { int o = gid - 256; g = g_cla[o]; be = be_cla[o]; m = m_cla[o]; v = v_cla[o]; }
    float s = g * rsqrtf(v + 1e-5f);
    bns[gid] = s;
    bnb[gid] = be - m * s;
  }
}

// ---------------------------------------------------------------------------
// Kernel 2: per-pixel 33-ch 1x1 conv (conf + offset + mask) + bilinear tables.
// Stores 4 clamped gather offsets + 4 mask-folded weights per tap; padded
// (zero) region handled by zeroing the weight.
// ---------------------------------------------------------------------------
__device__ __forceinline__ void tap_set(int r, int c, float g, float m,
                                        int& off, float& w) {
  int ri = r - 1, ci = c - 1;                 // padded -> interior coords
  bool ok = (ri >= 0) && (ri < 64) && (ci >= 0) && (ci < 64);
  off = ok ? (ri * 64 + ci) : 0;
  w   = ok ? g * m : 0.0f;
}

__global__ __launch_bounds__(256) void head_pre(
    const float* __restrict__ x,
    const float* __restrict__ w_conf, const float* __restrict__ b_conf,
    const float* __restrict__ w_om,   const float* __restrict__ b_om,
    float* __restrict__ out,
    int4* __restrict__ TabOff, float4* __restrict__ TabW)
{
  __shared__ float wl[256][36];               // [c][j], j<33, padded row
  for (int i = threadIdx.x; i < 256 * 33; i += 256) {
    int c = i / 33, j = i - c * 33;
    wl[c][j] = (j < 6) ? w_conf[j * 256 + c] : w_om[(j - 6) * 256 + c];
  }
  __syncthreads();

  const int pix = blockIdx.x * 256 + threadIdx.x;
  const int b = pix >> 12, hw = pix & 4095;
  const int h = hw >> 6, wcol = hw & 63;
  const float* xp = x + (size_t)b * C_ * HW_ + hw;

  float s[33];
#pragma unroll
  for (int j = 0; j < 33; ++j) s[j] = 0.0f;
  for (int c = 0; c < 256; ++c) {
    float xv = xp[(size_t)c * HW_];
    const float* wr = &wl[c][0];
#pragma unroll
    for (int j = 0; j < 33; ++j) s[j] += xv * wr[j];
  }

  // conf -> out[..][8 + j%2]
#pragma unroll
  for (int j = 0; j < 6; ++j) {
    int a = j >> 1, k = 8 + (j & 1);
    out[(((size_t)b * 3 + a) * 25 + k) * HW_ + hw] = s[j] + b_conf[j];
  }

  // bilinear tables per tap
#pragma unroll
  for (int n = 0; n < 9; ++n) {
    float o1 = s[6 + n]      + b_om[n];           // row offset
    float o2 = s[6 + 9 + n]  + b_om[9 + n];       // col offset
    float mk = s[6 + 18 + n] + b_om[18 + n];
    mk = 1.0f / (1.0f + __expf(-mk));             // sigmoid modulation
    float pr = (float)(h + 1 + (n / 3) - 1) + o1;
    float pc = (float)(wcol + 1 + (n % 3) - 1) + o2;
    pr = fminf(fmaxf(pr, 0.0f), 65.0f);
    pc = fminf(fmaxf(pc, 0.0f), 65.0f);
    float r0 = fminf(fmaxf(floorf(pr), 0.0f), 65.0f);
    float c0 = fminf(fmaxf(floorf(pc), 0.0f), 65.0f);
    float r1 = fminf(r0 + 1.0f, 65.0f);
    float c1 = fminf(c0 + 1.0f, 65.0f);
    float glt = (1.0f + (r0 - pr)) * (1.0f + (c0 - pc));
    float grb = (1.0f - (r1 - pr)) * (1.0f - (c1 - pc));
    float glb = (1.0f + (r0 - pr)) * (1.0f - (c1 - pc));
    float grt = (1.0f - (r1 - pr)) * (1.0f + (c0 - pc));
    int4 off; float4 wt;
    tap_set((int)r0, (int)c0, glt, mk, off.x, wt.x);
    tap_set((int)r1, (int)c1, grb, mk, off.y, wt.y);
    tap_set((int)r0, (int)c1, glb, mk, off.z, wt.z);
    tap_set((int)r1, (int)c0, grt, mk, off.w, wt.w);
    TabOff[(size_t)pix * 9 + n] = off;
    TabW [(size_t)pix * 9 + n] = wt;
  }
}

// ---------------------------------------------------------------------------
// Kernel 3: build bf16 im2col panel As[CHUNK_PX][2304] (shared by both
// branches). Gathers hit L2 (x = 32MB << 192MB L2).
// ---------------------------------------------------------------------------
__global__ __launch_bounds__(256) void sample_bf16(
    const float* __restrict__ x, const int4* __restrict__ TabOff,
    const float4* __restrict__ TabW, unsigned short* __restrict__ As,
    int chunkBase)
{
  __shared__ int4   toff[32 * 9];
  __shared__ float4 twt [32 * 9];
  const int t = threadIdx.x;
  const int pxb = blockIdx.x * 32;            // chunk-local pixel base
  for (int i = t; i < 288; i += 256) {
    size_t g = (size_t)(chunkBase + pxb + i / 9) * 9 + (i % 9);
    toff[i] = TabOff[g];
    twt [i] = TabW [g];
  }
  __syncthreads();

  const int pl  = t & 31;
  const int cst = t >> 5;                     // 0..7
  const int pixel = pxb + pl;                 // chunk-local
  const int gp = chunkBase + pixel;
  const float* xb = x + (size_t)(gp >> 12) * C_ * HW_;
  unsigned short* arow = As + (size_t)pixel * K_;

  for (int c = cst; c < 256; c += 8) {
    const float* xc = xb + (size_t)c * HW_;
#pragma unroll
    for (int n = 0; n < 9; ++n) {
      int4   o = toff[pl * 9 + n];
      float4 w = twt [pl * 9 + n];
      float v = w.x * xc[o.x] + w.y * xc[o.y] + w.z * xc[o.z] + w.w * xc[o.w];
      arow[c * 9 + n] = f2bf(v);
    }
  }
}

// ---------------------------------------------------------------------------
// Kernel 4: fused 512x2304xCHUNK_PX bf16 WMMA GEMM + BN + LeakyReLU.
// Block tile 128 outch x 128 px; 8 waves (4M x 2N); wave tile 32x64 =
// 2x4 v_wmma_f32_16x16x32_bf16 accumulators.
// ---------------------------------------------------------------------------
__global__ __launch_bounds__(256) void gemm_bf16(
    const unsigned short* __restrict__ Wp,    // [512][2304]
    const unsigned short* __restrict__ As,    // [CHUNK_PX][2304]
    const float* __restrict__ bns, const float* __restrict__ bnb,
    unsigned short* __restrict__ Act)         // [CHUNK_PX][512]
{
  __shared__ unsigned short Wl[128][40];      // padded stride: bank spread
  __shared__ unsigned short Al[128][40];

  const int t     = threadIdx.x;
  const int lane  = t & 31;
  const int wave  = t >> 5;
  const int waveM = wave & 3;                 // M offset = waveM*32
  const int waveN = wave >> 2;                // N offset = waveN*64
  const int lr    = lane & 15;
  const int lh    = lane >> 4;

  const int px0 = blockIdx.x * 128;           // chunk-local pixel tile
  const int oc0 = blockIdx.y * 128;           // output-channel tile

  const int lrow = t >> 1;                    // 0..127
  const int lcol = (t & 1) * 16;              // 0 or 16 (bf16 elems)

  const unsigned short* gw = Wp + (size_t)(oc0 + lrow) * K_ + lcol;
  const unsigned short* ga = As + (size_t)(px0 + lrow) * K_ + lcol;

  v8f zero = {};
  v8f acc[2][4];
#pragma unroll
  for (int m = 0; m < 2; ++m)
#pragma unroll
    for (int n = 0; n < 4; ++n) acc[m][n] = zero;

  union Frag { v16bf v; uint4 q[2]; };

  for (int kk = 0; kk < K_; kk += 32) {
    uint4 w0 = *(const uint4*)(gw);
    uint4 w1 = *(const uint4*)(gw + 8);
    uint4 a0 = *(const uint4*)(ga);
    uint4 a1 = *(const uint4*)(ga + 8);
    __builtin_prefetch(gw + 32, 0, 1);        // global_prefetch next K-slice
    __builtin_prefetch(ga + 32, 0, 1);
    gw += 32; ga += 32;

    __syncthreads();                          // previous tile fully consumed
    *(uint4*)&Wl[lrow][lcol]     = w0;
    *(uint4*)&Wl[lrow][lcol + 8] = w1;
    *(uint4*)&Al[lrow][lcol]     = a0;
    *(uint4*)&Al[lrow][lcol + 8] = a1;
    __syncthreads();

    Frag fa[2], fb[4];
#pragma unroll
    for (int m = 0; m < 2; ++m) {             // A: row=lane%16, 16 contig K per half
      const unsigned short* p = &Wl[waveM * 32 + m * 16 + lr][0];
      fa[m].q[0] = *(const uint4*)(p + lh * 16);
      fa[m].q[1] = *(const uint4*)(p + lh * 16 + 8);
    }
#pragma unroll
    for (int n = 0; n < 4; ++n) {             // B: col=lane%16, 16 contig K per half
      const unsigned short* p = &Al[waveN * 64 + n * 16 + lr][lh * 16];
      fb[n].q[0] = *(const uint4*)(p);
      fb[n].q[1] = *(const uint4*)(p + 8);
    }
#pragma unroll
    for (int m = 0; m < 2; ++m)
#pragma unroll
      for (int n = 0; n < 4; ++n)
        acc[m][n] = __builtin_amdgcn_wmma_f32_16x16x32_bf16(
            false, fa[m].v, false, fb[n].v, (short)0, acc[m][n], false, false);
  }

  // Fused BN + LeakyReLU epilogue -> bf16 activations [pixel][outch]
#pragma unroll
  for (int m = 0; m < 2; ++m) {
    const int ob = oc0 + waveM * 32 + m * 16 + lh * 8;  // D: M = j + 8*laneHalf
#pragma unroll
    for (int n = 0; n < 4; ++n) {
      const int px = px0 + waveN * 64 + n * 16 + lr;
      unsigned short* arow = Act + (size_t)px * OC_;
#pragma unroll
      for (int j = 0; j < 8; ++j) {
        const int o = ob + j;
        float v = acc[m][n][j] * bns[o] + bnb[o];
        v = (v >= 0.0f) ? v : 0.01f * v;
        arow[o] = f2bf(v);
      }
    }
  }
}

// ---------------------------------------------------------------------------
// Kernel 5: final 1x1 convs (24 loc + 45 cla outs, K=256 each) with
// interleaved (B,3,25,H,W) scatter into d_out.
// ---------------------------------------------------------------------------
__global__ __launch_bounds__(128) void head_post(
    const unsigned short* __restrict__ Act,
    const float* __restrict__ wlocx, const float* __restrict__ blocx,
    const float* __restrict__ wclax, const float* __restrict__ bclax,
    float* __restrict__ out, int chunkBase)
{
  __shared__ float arow[512];
  const int pl = blockIdx.x;                  // chunk-local pixel
  const int gp = chunkBase + pl;
  const int b = gp >> 12, hw = gp & 4095;
  const unsigned short* ap = Act + (size_t)pl * OC_;
  for (int i = threadIdx.x; i < 512; i += 128) arow[i] = bf2f(ap[i]);
  __syncthreads();

  const int t = threadIdx.x;
  if (t >= 69) return;
  const float* wp; float bias; int base, a, k;
  if (t < 24) { wp = wlocx + t * 256; bias = blocx[t]; base = 0;   a = t / 8;  k = t % 8; }
  else { int i = t - 24; wp = wclax + i * 256; bias = bclax[i]; base = 256; a = i / 15; k = 10 + i % 15; }

  float s = bias;
  for (int c = 0; c < 256; ++c) s += arow[base + c] * wp[c];
  out[(((size_t)b * 3 + a) * 25 + k) * (size_t)HW_ + hw] = s;
}

// ---------------------------------------------------------------------------
extern "C" void kernel_launch(void* const* d_in, const int* in_sizes, int n_in,
                              void* d_out, int out_size, void* d_ws, size_t ws_size,
                              hipStream_t stream) {
  const float* x      = (const float*)d_in[0];
  const float* w_conf = (const float*)d_in[1];
  const float* b_conf = (const float*)d_in[2];
  const float* w_om   = (const float*)d_in[3];
  const float* b_om   = (const float*)d_in[4];
  const float* w_dloc = (const float*)d_in[5];
  const float* g_loc  = (const float*)d_in[6];
  const float* be_loc = (const float*)d_in[7];
  const float* m_loc  = (const float*)d_in[8];
  const float* v_loc  = (const float*)d_in[9];
  const float* w_locx = (const float*)d_in[10];
  const float* b_locx = (const float*)d_in[11];
  const float* w_dcla = (const float*)d_in[12];
  const float* g_cla  = (const float*)d_in[13];
  const float* be_cla = (const float*)d_in[14];
  const float* m_cla  = (const float*)d_in[15];
  const float* v_cla  = (const float*)d_in[16];
  const float* w_clax = (const float*)d_in[17];
  const float* b_clax = (const float*)d_in[18];
  float* out = (float*)d_out;
  (void)in_sizes; (void)n_in; (void)out_size; (void)ws_size;

  char* ws = (char*)d_ws;
  size_t off = 0;
  auto take = [&](size_t bytes) -> char* {
    char* p = ws + off;
    off = (off + bytes + 255) & ~(size_t)255;
    return p;
  };
  int4*           TabOff = (int4*)  take((size_t)NP_ * 9 * sizeof(int4));   // 4.7 MB
  float4*         TabW   = (float4*)take((size_t)NP_ * 9 * sizeof(float4)); // 4.7 MB
  unsigned short* Wp     = (unsigned short*)take((size_t)OC_ * K_ * 2);     // 2.4 MB
  float*          bns    = (float*) take((size_t)OC_ * 4);
  float*          bnb    = (float*) take((size_t)OC_ * 4);
  unsigned short* As     = (unsigned short*)take((size_t)CHUNK_PX * K_ * 2);   // 37.7 MB
  unsigned short* Act    = (unsigned short*)take((size_t)CHUNK_PX * OC_ * 2);  // 8.4 MB

  pack_prep<<<(OC_ * K_ + 255) / 256, 256, 0, stream>>>(
      w_dloc, w_dcla, g_loc, be_loc, m_loc, v_loc,
      g_cla, be_cla, m_cla, v_cla, Wp, bns, bnb);

  head_pre<<<NP_ / 256, 256, 0, stream>>>(
      x, w_conf, b_conf, w_om, b_om, out, TabOff, TabW);

  for (int ch = 0; ch < NCHUNK; ++ch) {
    const int cb = ch * CHUNK_PX;
    sample_bf16<<<CHUNK_PX / 32, 256, 0, stream>>>(x, TabOff, TabW, As, cb);
    dim3 g(CHUNK_PX / 128, OC_ / 128);        // 64 x 4 = 256 workgroups
    gemm_bf16<<<g, 256, 0, stream>>>(Wp, As, bns, bnb, Act);
    head_post<<<CHUNK_PX, 128, 0, stream>>>(Act, w_locx, b_locx, w_clax, b_clax, out, cb);
  }
}
